// NetConvolve_15779709846105
// MI455X (gfx1250) — matrix operand order repaired and checked by reading, not compiled, not run on hardware
//
#include <hip/hip_runtime.h>

// NetConvolve on gfx1250: sliding-window 1D conv == GEMM with K = KSIZE*C = 32,
// mapped onto V_WMMA_F32_16X16X32_F16 (f16 A/B, fp32 accumulate).
// Output-bandwidth bound (258.5 MB stores -> ~11us floor @ 23.3 TB/s);
// f16 WMMA keeps compute (~4.1 GFLOP) far below that floor.
// Each wave computes 4 consecutive 16x32 output tiles of one slice, so the
// constant B matrix (W as 32x32) and bias are gathered once per 8 WMMAs.

typedef __attribute__((ext_vector_type(16))) _Float16 v16h;
typedef __attribute__((ext_vector_type(8)))  float    v8f;
typedef __attribute__((ext_vector_type(2)))  float    v2f;

#define B_      32
#define N_      32768
#define STRIDE_ 256
#define KS_     16
#define F_      32
#define NSLICE_ 127
#define OUTLEN_ 497                     // WINDOW - KSIZE + 1
#define TILES_  32                      // ceil(OUTLEN/16)
#define GROUPS_ 8                       // 4 tiles per wave -> 8 groups/slice
#define OROWS_  (NSLICE_ * OUTLEN_)     // 63119 output rows per batch

__global__ __launch_bounds__(256)
void conv_wmma_f16_kernel(const float* __restrict__ x,
                          const float* __restrict__ W,
                          const float* __restrict__ bias,
                          float* __restrict__ out)
{
    const int lane  = threadIdx.x & 31;      // wave32
    const int wave  = blockIdx.x * 8 + (threadIdx.x >> 5);

    const int group = wave & (GROUPS_ - 1);  // 4-tile group within a slice
    const int rest  = wave >> 3;
    const int slice = rest % NSLICE_;
    const int batch = rest / NSLICE_;

    const int M      = lane & 15;            // row (A/D) or column (B/D) index
    const int laneHi = lane >> 4;            // 0 = lanes 0-15, 1 = lanes 16-31

    // ---------------- B matrices (once per wave): W as 32x32, two N-tiles ---
    // B k order used here (same as A): lane n, element i -> K = 16*laneHi + i, N = n%16.
    const int kb = laneHi << 4;
    v16h bm0, bm1;
#pragma unroll
    for (int i = 0; i < 16; ++i) {
        bm0[i] = (_Float16)W[(kb + i) * F_ + M];
        bm1[i] = (_Float16)W[(kb + i) * F_ + 16 + M];
    }
    const float bv0 = bias[M];               // f32 C/D layout: col = lane%16
    const float bv1 = bias[16 + M];

    const float* xb    = x + (size_t)batch * N_ * 2;
    const size_t obase = ((size_t)batch * OROWS_ + (size_t)slice * OUTLEN_) * F_;
    const int    pmx   = N_ - KS_;

#pragma unroll
    for (int j = 0; j < 4; ++j) {
        const int t0 = group * 64 + j * 16;  // tile start (output position)

        // ---- A: 16 output rows x 32 (k,c). Row t's contraction vector is 32
        // contiguous floats at x[b, slice*256+t, 0]; in the k order used here (same as B),
        // lane L's v16h = 16 contiguous floats starting at row + 16*laneHi.
        int pos = slice * STRIDE_ + t0 + M;
        pos = (pos > pmx) ? pmx : pos;       // clamp: tail rows discarded at store
        const v2f* rp2 = (const v2f*)(xb + (size_t)pos * 2) + (laneHi << 3);

        v16h a;
#pragma unroll
        for (int q = 0; q < 8; ++q) {
            v2f d = rp2[q];
            a[2 * q]     = (_Float16)d.x;
            a[2 * q + 1] = (_Float16)d.y;
        }

        v8f c0, c1;
#pragma unroll
        for (int r = 0; r < 8; ++r) { c0[r] = bv0; c1[r] = bv1; }

        // ---- single-K WMMAs (K = 32 exactly) ----
        c0 = __builtin_amdgcn_wmma_f32_16x16x32_f16(false, a, false, bm0,
                                                    (short)0, c0, false, false);
        c1 = __builtin_amdgcn_wmma_f32_16x16x32_f16(false, a, false, bm1,
                                                    (short)0, c1, false, false);

        // ---- ReLU + store (D layout: row = r + 8*laneHi, col = lane%16) ----
        const int trow0 = t0 + laneHi * 8;
        float* op = out + obase + (size_t)trow0 * F_;

        if (t0 + 16 <= OUTLEN_) {            // wave-uniform: all tiles but the tail
#pragma unroll
            for (int r = 0; r < 8; ++r) {
                const float v0 = c0[r];
                const float v1 = c1[r];
                op[(size_t)r * F_ + M]      = v0 > 0.f ? v0 : 0.f;
                op[(size_t)r * F_ + 16 + M] = v1 > 0.f ? v1 : 0.f;
            }
        } else {                             // tail tile: 497 = 31*16 + 1
#pragma unroll
            for (int r = 0; r < 8; ++r) {
                if (trow0 + r < OUTLEN_) {
                    const float v0 = c0[r];
                    const float v1 = c1[r];
                    op[(size_t)r * F_ + M]      = v0 > 0.f ? v0 : 0.f;
                    op[(size_t)r * F_ + 16 + M] = v1 > 0.f ? v1 : 0.f;
                }
            }
        }
    }
}

extern "C" void kernel_launch(void* const* d_in, const int* in_sizes, int n_in,
                              void* d_out, int out_size, void* d_ws, size_t ws_size,
                              hipStream_t stream) {
    (void)in_sizes; (void)n_in; (void)out_size; (void)d_ws; (void)ws_size;
    const float* x  = (const float*)d_in[0];
    const float* W  = (const float*)d_in[1];
    const float* b  = (const float*)d_in[2];
    float* out      = (float*)d_out;

    // 32 batches * 127 slices * 8 groups = 32512 waves; 8 waves (256 thr) / block.
    const int blocks = (B_ * NSLICE_ * GROUPS_) / 8;   // 4064 — exact, EXEC all-ones
    conv_wmma_f16_kernel<<<blocks, 256, 0, stream>>>(x, W, b, out);
}
